// Attn_51539607676
// MI455X (gfx1250) — matrix operand order repaired and checked by reading, not compiled, not run on hardware
//
#include <hip/hip_runtime.h>
#include <hip/hip_bf16.h>

// ---------- types ----------
typedef __attribute__((ext_vector_type(16))) _Float16 v16h;
typedef __attribute__((ext_vector_type(8)))  _Float16 v8h;
typedef __attribute__((ext_vector_type(8)))  float    v8f;

#define WMMA(a, b, c) \
  __builtin_amdgcn_wmma_f32_16x16x32_f16(false, (a), false, (b), (short)0, (c), false, false)

// ---------- WMMA fragment loaders (contiguous k order, same in A and B) ----------
// A fragment 16x32 f16: lane<16 -> row r, K = 0..15; lane>=16 -> K = 16..31 (contiguous)
__device__ __forceinline__ v16h load_frag_a(const _Float16* __restrict__ base, int ld,
                                            int row, int k, int lane) {
  const int r = lane & 15, hs = lane >> 4;
  const _Float16* p = base + (size_t)(row + r) * ld + k + hs * 16;
  v8h lo = *(const v8h*)(p);
  v8h hi = *(const v8h*)(p + 8);
  v16h out;
#pragma unroll
  for (int i = 0; i < 8; i++) { out[i] = lo[i]; out[i + 8] = hi[i]; }
  return out;
}

// B fragment 32x16 f16 with B[k,n] = src[col0+n][k] (row-major source):
// lane<16 -> column n=lane, K = k..k+15; lane>=16 -> K = k+16..k+31 (contiguous)
__device__ __forceinline__ v16h load_frag_b(const _Float16* __restrict__ base, int ld,
                                            int col, int k, int lane) {
  const int r = lane & 15, hs = lane >> 4;
  const _Float16* p = base + (size_t)(col + r) * ld + k + hs * 16;
  v8h lo = *(const v8h*)(p);
  v8h hi = *(const v8h*)(p + 8);
  v16h out;
#pragma unroll
  for (int i = 0; i < 8; i++) { out[i] = lo[i]; out[i + 8] = hi[i]; }
  return out;
}

// ---------- elementwise kernels ----------
__global__ void cvt_f32_to_f16(const float* __restrict__ in, _Float16* __restrict__ out, int n) {
  int i = blockIdx.x * blockDim.x + threadIdx.x;
  if (i < n) out[i] = (_Float16)in[i];
}

// P [B,S,H*dh] f32 -> out [B*H, S, dh] f16 with RoPE; scale folds 1/sqrt(dh) into Q
__global__ void rope_to_heads(const float* __restrict__ P, _Float16* __restrict__ out,
                              int B, int S, int H, int dh, float scale, float theta) {
  int idx = blockIdx.x * blockDim.x + threadIdx.x;
  const int half = dh >> 1;
  int total = B * S * H * half;
  if (idx >= total) return;
  int i = idx % half;
  int h = (idx / half) % H;
  int s = (idx / (half * H)) % S;
  int b =  idx / (half * H * S);
  float fr  = powf(theta, -(2.0f * (float)i) / (float)dh);
  float ang = (float)s * fr;
  float cs = cosf(ang), sn = sinf(ang);
  const float* pp = P + ((size_t)(b * S + s)) * (size_t)(H * dh) + h * dh + 2 * i;
  float x1 = pp[0], x2 = pp[1];
  _Float16* op = out + (((size_t)(b * H + h)) * S + s) * dh + 2 * i;
  op[0] = (_Float16)((x1 * cs - x2 * sn) * scale);
  op[1] = (_Float16)((x1 * sn + x2 * cs) * scale);
}

// P [B,S,H*dh] f32 -> Vt [B*H, dh, S] f16 (transposed so PV A-fragments load contiguously)
__global__ void v_to_vt(const float* __restrict__ P, _Float16* __restrict__ Vt,
                        int B, int S, int H, int dh) {
  int idx = blockIdx.x * blockDim.x + threadIdx.x;
  int total = B * H * dh * S;
  if (idx >= total) return;
  int s = idx % S;
  int d = (idx / S) % dh;
  int h = (idx / (S * dh)) % H;
  int b =  idx / (S * dh * H);
  float v = P[((size_t)(b * S + s)) * (size_t)(H * dh) + h * dh + d];
  Vt[(((size_t)(b * H + h)) * dh + d) * S + s] = (_Float16)v;
}

// ---------- GEMM: C[m,n] = sum_k A[m,k] * W[n,k]  (X @ W^T), f16 in, f32 out ----------
__global__ void __launch_bounds__(256) gemm_nt_f16f32(const _Float16* __restrict__ A,
                                                      const _Float16* __restrict__ W,
                                                      float* __restrict__ C,
                                                      int M, int N, int K) {
  const int lane = threadIdx.x & 31;
  const int w    = threadIdx.x >> 5;                 // 8 waves per block
  const int m0 = blockIdx.x * 128 + (w >> 1) * 32;   // 4 waves along M
  const int n0 = blockIdx.y * 128 + (w & 1)  * 64;   // 2 waves along N
  v8f acc[2][4];
#pragma unroll
  for (int i = 0; i < 2; i++)
#pragma unroll
    for (int j = 0; j < 4; j++) acc[i][j] = (v8f)0.0f;

  for (int k = 0; k < K; k += 32) {
    v16h a0 = load_frag_a(A, K, m0,      k, lane);
    v16h a1 = load_frag_a(A, K, m0 + 16, k, lane);
    v16h b0 = load_frag_b(W, K, n0,      k, lane);
    v16h b1 = load_frag_b(W, K, n0 + 16, k, lane);
    v16h b2 = load_frag_b(W, K, n0 + 32, k, lane);
    v16h b3 = load_frag_b(W, K, n0 + 48, k, lane);
    acc[0][0] = WMMA(a0, b0, acc[0][0]);
    acc[0][1] = WMMA(a0, b1, acc[0][1]);
    acc[0][2] = WMMA(a0, b2, acc[0][2]);
    acc[0][3] = WMMA(a0, b3, acc[0][3]);
    acc[1][0] = WMMA(a1, b0, acc[1][0]);
    acc[1][1] = WMMA(a1, b1, acc[1][1]);
    acc[1][2] = WMMA(a1, b2, acc[1][2]);
    acc[1][3] = WMMA(a1, b3, acc[1][3]);
  }
  const int r = lane & 15, hs = lane >> 4;
#pragma unroll
  for (int i = 0; i < 2; i++)
#pragma unroll
    for (int j = 0; j < 4; j++) {
      float* cp = C + (size_t)(m0 + 16 * i + hs * 8) * N + n0 + 16 * j + r;
#pragma unroll
      for (int e = 0; e < 8; e++) cp[(size_t)e * N] = acc[i][j][e];
    }
}

// ---------- causal flash attention, transposed-score formulation ----------
// Q [BH,S,64] f16 (pre-scaled by 1/8 + RoPE), K [BH,S,64] f16 (RoPE), Vt [BH,64,S] f16
// O [B,S,H*64] f16. One wave per 16-query tile; S^T = K*Q^T so softmax stats are per-lane
// (q along lanes, k along VGPRs) with only a lane<->lane+16 shfl_xor exchange.
__global__ void __launch_bounds__(128) flash_attn(const _Float16* __restrict__ Q,
                                                  const _Float16* __restrict__ Kx,
                                                  const _Float16* __restrict__ Vt,
                                                  _Float16* __restrict__ O,
                                                  int S, int H) {
  const int lane = threadIdx.x & 31;
  const int wv   = threadIdx.x >> 5;           // 4 waves / block
  const int bh   = blockIdx.y;
  const int q0   = (blockIdx.x * 4 + wv) * 16;
  const int r = lane & 15, hs = lane >> 4;
  const int qcol = q0 + r;                     // this lane's query row

  const _Float16* Qb = Q  + (size_t)bh * S * 64;
  const _Float16* Kb = Kx + (size_t)bh * S * 64;
  const _Float16* Vb = Vt + (size_t)bh * 64 * S;

  // Q^T B-fragments (dh chunks 0..31 / 32..63), loaded once
  v16h bq0 = load_frag_b(Qb, 64, q0, 0,  lane);
  v16h bq1 = load_frag_b(Qb, 64, q0, 32, lane);

  v8f o0 = (v8f)0.0f, o1 = (v8f)0.0f, o2 = (v8f)0.0f, o3 = (v8f)0.0f; // O^T, d chunks
  float m_run = -1e30f, l_run = 0.0f;

  const int jend = q0 + 16;                    // causal: k <= q0+15
  for (int j = 0; j < jend; j += 32) {
    // scores^T tiles: s0 = k rows j..j+15, s1 = j+16..j+31 (C layout: k in VGPRs, q in lanes)
    v8f s0 = (v8f)0.0f, s1 = (v8f)0.0f;
    v16h ak;
    ak = load_frag_a(Kb, 64, j,      0,  lane); s0 = WMMA(ak, bq0, s0);
    ak = load_frag_a(Kb, 64, j,      32, lane); s0 = WMMA(ak, bq1, s0);
    ak = load_frag_a(Kb, 64, j + 16, 0,  lane); s1 = WMMA(ak, bq0, s1);
    ak = load_frag_a(Kb, 64, j + 16, 32, lane); s1 = WMMA(ak, bq1, s1);

    const int k0 = j + hs * 8, k1 = j + 16 + hs * 8;
    float p0[8], p1[8];
    float mloc = -1e30f;
#pragma unroll
    for (int e = 0; e < 8; e++) {
      float a = (k0 + e <= qcol) ? s0[e] : -1e30f;   // causal mask
      float b = (k1 + e <= qcol) ? s1[e] : -1e30f;
      p0[e] = a; p1[e] = b;
      mloc = fmaxf(mloc, fmaxf(a, b));
    }
    // combine running max across the two lane-halves that share this q column
    float mo   = __shfl_xor(mloc, 16, 32);
    float mnew = fmaxf(m_run, fmaxf(mloc, mo));
    float sc   = __expf(m_run - mnew);
    m_run = mnew;
#pragma unroll
    for (int e = 0; e < 8; e++) { o0[e] *= sc; o1[e] *= sc; o2[e] *= sc; o3[e] *= sc; }
    l_run *= sc;

    float lsum = 0.0f;
#pragma unroll
    for (int e = 0; e < 8; e++) {
      p0[e] = __expf(p0[e] - mnew);
      p1[e] = __expf(p1[e] - mnew);
      lsum += p0[e] + p1[e];
    }
    l_run += lsum;

    // pack P^T into a B fragment (32 k x 16 q): lanes<16 need tile0 rows 0..15,
    // lanes>=16 need tile1 rows 0..15 -> complementary halves come from lane^16
    v16h bp;
#pragma unroll
    for (int e = 0; e < 8; e++) {
      float x0 = __shfl_xor(p0[e], 16, 32);
      float x1 = __shfl_xor(p1[e], 16, 32);
      bp[e]     = (_Float16)((hs == 0) ? p0[e] : x1);
      bp[e + 8] = (_Float16)((hs == 0) ? x0    : p1[e]);
    }

    // O^T += V^T * P^T (A = Vt rows = d, K-dim = k rows j..j+31)
    v16h av;
    av = load_frag_a(Vb, S, 0,  j, lane); o0 = WMMA(av, bp, o0);
    av = load_frag_a(Vb, S, 16, j, lane); o1 = WMMA(av, bp, o1);
    av = load_frag_a(Vb, S, 32, j, lane); o2 = WMMA(av, bp, o2);
    av = load_frag_a(Vb, S, 48, j, lane); o3 = WMMA(av, bp, o3);
  }

  float ltot = l_run + __shfl_xor(l_run, 16, 32);
  float inv  = (ltot > 0.0f) ? (1.0f / ltot) : 0.0f;

  // store O^T fragment -> O [B,S,H*dh] f16; lane holds q=qcol, d = dc*16 + hs*8 + e
  const int b = bh / H, h = bh % H;
  _Float16* orow = O + ((size_t)(b * S + qcol)) * (size_t)(H * 64) + h * 64 + hs * 8;
#pragma unroll
  for (int e = 0; e < 8; e++) {
    orow[e]      = (_Float16)(o0[e] * inv);
    orow[16 + e] = (_Float16)(o1[e] * inv);
    orow[32 + e] = (_Float16)(o2[e] * inv);
    orow[48 + e] = (_Float16)(o3[e] * inv);
  }
}

// ---------- launcher ----------
extern "C" void kernel_launch(void* const* d_in, const int* in_sizes, int n_in,
                              void* d_out, int out_size, void* d_ws, size_t ws_size,
                              hipStream_t stream) {
  const float* x  = (const float*)d_in[0];
  const float* Wq = (const float*)d_in[1];
  const float* Wk = (const float*)d_in[2];
  const float* Wv = (const float*)d_in[3];
  const float* Wo = (const float*)d_in[4];
  (void)in_sizes; (void)n_in; (void)out_size;

  const int B = 4, S = 2048, D = 1024, H = 16, dh = 64;
  const int M  = B * S;           // 8192
  const int NX = M * D;           // 8388608 elements of x / each activation
  const int NW = D * D;           // 1048576 per weight

  // workspace layout (120 MB total)
  _Float16* xh  = (_Float16*)d_ws;          // 16 MB
  _Float16* wqh = xh  + NX;                 // 2 MB
  _Float16* wkh = wqh + NW;
  _Float16* wvh = wkh + NW;
  _Float16* woh = wvh + NW;
  float*    pf  = (float*)(woh + NW);       // 32 MB fp32 projection scratch (reused)
  _Float16* qh  = (_Float16*)(pf + NX);     // 16 MB [BH,S,dh]
  _Float16* kh  = qh  + NX;                 // 16 MB [BH,S,dh]
  _Float16* vth = kh  + NX;                 // 16 MB [BH,dh,S]
  _Float16* oh  = vth + NX;                 // 16 MB [B,S,D]
  (void)ws_size;

  const int T = 256;
  // f32 -> f16 conversions
  cvt_f32_to_f16<<<(NX + T - 1) / T, T, 0, stream>>>(x,  xh,  NX);
  cvt_f32_to_f16<<<(NW + T - 1) / T, T, 0, stream>>>(Wq, wqh, NW);
  cvt_f32_to_f16<<<(NW + T - 1) / T, T, 0, stream>>>(Wk, wkh, NW);
  cvt_f32_to_f16<<<(NW + T - 1) / T, T, 0, stream>>>(Wv, wvh, NW);
  cvt_f32_to_f16<<<(NW + T - 1) / T, T, 0, stream>>>(Wo, woh, NW);

  dim3 gg(M / 128, D / 128);     // (64, 8)
  const int nrope = B * S * H * (dh / 2);   // 4194304
  const int ntr   = B * H * dh * S;         // 8388608

  // Q = x @ Wq^T -> RoPE (+ 1/sqrt(dh) folded in) -> qh
  gemm_nt_f16f32<<<gg, 256, 0, stream>>>(xh, wqh, pf, M, D, D);
  rope_to_heads<<<(nrope + T - 1) / T, T, 0, stream>>>(pf, qh, B, S, H, dh, 0.125f, 10000.0f);
  // K = x @ Wk^T -> RoPE -> kh
  gemm_nt_f16f32<<<gg, 256, 0, stream>>>(xh, wkh, pf, M, D, D);
  rope_to_heads<<<(nrope + T - 1) / T, T, 0, stream>>>(pf, kh, B, S, H, dh, 1.0f, 10000.0f);
  // V = x @ Wv^T -> transpose -> vth
  gemm_nt_f16f32<<<gg, 256, 0, stream>>>(xh, wvh, pf, M, D, D);
  v_to_vt<<<(ntr + T - 1) / T, T, 0, stream>>>(pf, vth, B, S, H, dh);

  // causal flash attention
  flash_attn<<<dim3(S / 64, B * H), 128, 0, stream>>>(qh, kh, vth, oh, S, H);

  // final projection: out = O @ Wo^T (fp32 to d_out)
  gemm_nt_f16f32<<<gg, 256, 0, stream>>>(oh, woh, (float*)d_out, M, D, D);
}
